// PolymerGNN_v12_Res_25950192403181
// MI455X (gfx1250) — matrix operand order repaired and checked by reading, not compiled, not run on hardware
//
#include <hip/hip_runtime.h>
#include <hip/hip_bf16.h>

// ---------------------------------------------------------------------------
// PolymerGNN (4x GATv2 + BN/ELU + mean-pool + 5 task heads) for MI455X gfx1250
//
//  * Node projections -> tiled WMMA GEMM (v_wmma_f32_16x16x32_bf16), bf16 in /
//    f32 accum. GEMM total ~26 GF -> negligible vs ~1 GB/layer of random edge
//    gather/scatter at 23.3 TB/s; bf16 is the right precision/perf point.
//  * A tile staged row-major, B tile staged TRANSPOSED (column-major) in LDS
//    so each 16-half WMMA fragment is two contiguous 8-half K runs -> ds_load_b128 pairs.
//    Staging uses float4 global loads (global_load_b128) packed into v4bf
//    ds_store_b64, plus global_prefetch_b8 of the next K-tile.
//  * Edge stage: one wave32 per edge; lane d owns channel 32h+d -> coalesced
//    512B row gathers; lin_edge recomputed from 8KB LDS copy of We.
//  * Segment softmax: order-preserving-uint atomicMax + f32 atomicAdd.
//  * BatchNorm(train stats)+ELU fused; bias folded into stats math.
// ---------------------------------------------------------------------------

#define NNODES 100000
#define NEDGES 400000
#define FNODE  64
#define FEDGE  16
#define HIDDIM 128
#define NHEADS 4
#define DHEAD  32
#define NLAYER 4
#define NGRAPH 1024
#define FPBITS 2048
#define NTASK  5
#define KSEL   32

typedef __attribute__((ext_vector_type(16))) __bf16 v16bf;
typedef __attribute__((ext_vector_type(8)))  __bf16 v8bf;
typedef __attribute__((ext_vector_type(4)))  __bf16 v4bf;
typedef __attribute__((ext_vector_type(8)))  float  v8f;

// ---- order-preserving float<->uint encoding for atomicMax softmax-max ----
__device__ __forceinline__ unsigned enc_f32(float f) {
    unsigned u = __float_as_uint(f);
    return (u & 0x80000000u) ? ~u : (u | 0x80000000u);
}
__device__ __forceinline__ float dec_f32(unsigned u) {
    return __uint_as_float((u & 0x80000000u) ? (u ^ 0x80000000u) : ~u);
}

// ===========================================================================
// Tiled GEMM  C[M,128] = A[M,K] @ W[K,128] + bias  (bf16 WMMA, f32 accum)
// Block = 256 threads = 8 waves; tile 128x128; waves in 4(M) x 2(N) grid,
// each wave computes a 32x64 region = 2x4 WMMA 16x16 tiles. K % 32 == 0.
// ===========================================================================
#define A_STRIDE 40   // 128 rows x 32 K-halves (+8 pad)  : As[r][k]
#define B_STRIDE 40   // 128 cols x 32 K-halves (+8 pad)  : Bst[c][k] (transposed)

__global__ void __launch_bounds__(256)
gemm_bf16_wmma_kernel(const float* __restrict__ A, const float* __restrict__ W,
                      const float* __restrict__ bias, float* __restrict__ C,
                      int M, int K) {
    __shared__ __bf16 As [128 * A_STRIDE];
    __shared__ __bf16 Bst[128 * B_STRIDE];

    const int lane  = threadIdx.x & 31;
    const int wave  = threadIdx.x >> 5;
    const int wm    = wave >> 1;           // 0..3 : 32-row band
    const int wn    = wave & 1;            // 0..1 : 64-col band
    const int lhalf = lane & 15;
    const int ahi   = (lane >> 4) << 3;    // A frag: +8  K for lanes 16..31
    const int bhi   = (lane >> 4) << 3;    // B frag: +8  K for lanes 16..31
    const int rowBase = blockIdx.x * 128;
    const bool fullTile = (rowBase + 128 <= M);

    v8f acc[2][4];
    #pragma unroll
    for (int mi = 0; mi < 2; ++mi)
        #pragma unroll
        for (int ni = 0; ni < 4; ++ni) acc[mi][ni] = (v8f){0,0,0,0,0,0,0,0};

    for (int k0 = 0; k0 < K; k0 += 32) {
        const bool pf = (k0 + 32 < K);
        // ---- stage A tile (128x32): float4 loads -> v4bf b64 stores ----
        #pragma unroll
        for (int t = threadIdx.x; t < 1024; t += 256) {
            int r  = t >> 3;             // 0..127
            int c4 = (t & 7) << 2;       // 0,4,...,28
            int gr = rowBase + r;
            float4 v = make_float4(0.f, 0.f, 0.f, 0.f);
            if (fullTile || gr < M) {
                v = *(const float4*)&A[(size_t)gr * K + (k0 + c4)];
                if (pf)   // CDNA5 global_prefetch_b8 of next K-tile
                    __builtin_prefetch(&A[(size_t)gr * K + (k0 + 32 + c4)], 0, 1);
            }
            v4bf p = { (__bf16)v.x, (__bf16)v.y, (__bf16)v.z, (__bf16)v.w };
            *(v4bf*)&As[r * A_STRIDE + c4] = p;
        }
        // ---- stage B tile (32x128) TRANSPOSED: Bst[c][k] ----
        #pragma unroll
        for (int t = threadIdx.x; t < 1024; t += 256) {
            int c  = t & 127;            // column (coalesced global reads)
            int rg = (t >> 7) << 2;      // K-group base: 0,4,...,28
            v4bf p;
            #pragma unroll
            for (int q = 0; q < 4; ++q)
                p[q] = (__bf16)W[(size_t)(k0 + rg + q) * HIDDIM + c];
            *(v4bf*)&Bst[c * B_STRIDE + rg] = p;
        }
        __syncthreads();

        // ---- fragments: contiguous 8-half runs -> ds_load_b128 ----
        v16bf afrag[2], bfrag[4];
        union Pack16 { v16bf v; v8bf h[2]; };
        #pragma unroll
        for (int mi = 0; mi < 2; ++mi) {
            int r = wm * 32 + mi * 16 + lhalf;
            Pack16 u;
            u.h[0] = *(const v8bf*)&As[r * A_STRIDE + ahi];        // K ahi..ahi+7
            u.h[1] = *(const v8bf*)&As[r * A_STRIDE + 16 + ahi];   // K 16+ahi..
            afrag[mi] = u.v;
        }
        #pragma unroll
        for (int ni = 0; ni < 4; ++ni) {
            int c = wn * 64 + ni * 16 + lhalf;
            Pack16 u;
            u.h[0] = *(const v8bf*)&Bst[c * B_STRIDE + bhi];       // K bhi..bhi+7
            u.h[1] = *(const v8bf*)&Bst[c * B_STRIDE + 16 + bhi];  // K 16+bhi..
            bfrag[ni] = u.v;
        }
        #pragma unroll
        for (int mi = 0; mi < 2; ++mi)
            #pragma unroll
            for (int ni = 0; ni < 4; ++ni)
                acc[mi][ni] = __builtin_amdgcn_wmma_f32_16x16x32_bf16(
                    false, afrag[mi], false, bfrag[ni],
                    (short)0, acc[mi][ni], false, false);
        __syncthreads();
    }

    // ---- store D (+bias); C/D layout: VGPR r -> M = r + 8*(lane>=16) ----
    const int mo = (lane >> 4) << 3;
    #pragma unroll
    for (int ni = 0; ni < 4; ++ni) {
        int col = wn * 64 + ni * 16 + lhalf;
        float bv = bias[col];
        #pragma unroll
        for (int mi = 0; mi < 2; ++mi) {
            #pragma unroll
            for (int r2 = 0; r2 < 8; ++r2) {
                int row = rowBase + wm * 32 + mi * 16 + r2 + mo;
                if (fullTile || row < M)
                    C[(size_t)row * HIDDIM + col] = acc[mi][ni][r2] + bv;
            }
        }
    }
}

// ===========================================================================
// Edge pass 1: score[e,h] = att_h . leaky_relu(xl[src]+xr[dst]+ea@We);
// atomicMax of encoded score into m_u[dst,h].  One wave32 per edge.
// ===========================================================================
__global__ void __launch_bounds__(256)
edge_score_kernel(const int* __restrict__ src, const int* __restrict__ dst,
                  const float* __restrict__ ea, const float* __restrict__ We,
                  const float* __restrict__ att, const float* __restrict__ xl,
                  const float* __restrict__ xr, float* __restrict__ score,
                  unsigned* __restrict__ m_u) {
    __shared__ float WeS[FEDGE * HIDDIM];
    __shared__ float attS[HIDDIM];
    for (int i = threadIdx.x; i < FEDGE * HIDDIM; i += 256) WeS[i] = We[i];
    for (int i = threadIdx.x; i < HIDDIM; i += 256) attS[i] = att[i];
    __syncthreads();

    int e = blockIdx.x * 8 + (threadIdx.x >> 5);
    if (e >= NEDGES) return;
    int lane = threadIdx.x & 31;
    int s = src[e], d = dst[e];

    float eav[FEDGE];
    #pragma unroll
    for (int k = 0; k < FEDGE; ++k) eav[k] = ea[(size_t)e * FEDGE + k];

    float sc[NHEADS];
    #pragma unroll
    for (int h = 0; h < NHEADS; ++h) {
        int c = h * DHEAD + lane;
        float ee = 0.0f;
        #pragma unroll
        for (int k = 0; k < FEDGE; ++k) ee += eav[k] * WeS[k * HIDDIM + c];
        float v = xl[(size_t)s * HIDDIM + c] + xr[(size_t)d * HIDDIM + c] + ee;
        v = (v > 0.0f) ? v : 0.2f * v;                 // leaky_relu(0.2)
        float prod = v * attS[c];
        #pragma unroll
        for (int o = 16; o > 0; o >>= 1) prod += __shfl_xor(prod, o, 32);
        sc[h] = prod;
    }
    if (lane == 0) {
        #pragma unroll
        for (int h = 0; h < NHEADS; ++h) {
            score[(size_t)e * NHEADS + h] = sc[h];
            atomicMax(&m_u[(size_t)d * NHEADS + h], enc_f32(sc[h]));
        }
    }
}

// Edge pass 2: p = exp(score - max[dst]); denom[dst] += p. Thread per (e,h).
__global__ void edge_exp_kernel(const int* __restrict__ dst,
                                const unsigned* __restrict__ m_u,
                                float* __restrict__ score,
                                float* __restrict__ denom) {
    int i = blockIdx.x * blockDim.x + threadIdx.x;
    if (i >= NEDGES * NHEADS) return;
    int e = i >> 2, h = i & 3;
    int d = dst[e];
    float p = __expf(score[i] - dec_f32(m_u[(size_t)d * NHEADS + h]));
    score[i] = p;
    atomicAdd(&denom[(size_t)d * NHEADS + h], p);
}

// Edge pass 3: acc[dst,c] += (p/denom[dst]) * xl[src,c]. One wave32 per edge.
__global__ void __launch_bounds__(256)
edge_scatter_kernel(const int* __restrict__ src, const int* __restrict__ dst,
                    const float* __restrict__ score, const float* __restrict__ denom,
                    const float* __restrict__ xl, float* __restrict__ acc) {
    int e = blockIdx.x * 8 + (threadIdx.x >> 5);
    if (e >= NEDGES) return;
    int lane = threadIdx.x & 31;
    int s = src[e], d = dst[e];
    float w[NHEADS];
    #pragma unroll
    for (int h = 0; h < NHEADS; ++h)
        w[h] = score[(size_t)e * NHEADS + h] /
               (denom[(size_t)d * NHEADS + h] + 1e-16f);
    #pragma unroll
    for (int h = 0; h < NHEADS; ++h) {
        int c = h * DHEAD + lane;
        atomicAdd(&acc[(size_t)d * HIDDIM + c], w[h] * xl[(size_t)s * HIDDIM + c]);
    }
}

// ===========================================================================
// BatchNorm stats (per-channel sum / sumsq over N), bias folded in later.
// stride is a multiple of 128 -> each thread owns one fixed channel.
// ===========================================================================
__global__ void __launch_bounds__(256)
bn_stats_kernel(const float* __restrict__ acc, float* __restrict__ sums,
                float* __restrict__ sumsq) {
    size_t tid = blockIdx.x * (size_t)blockDim.x + threadIdx.x;
    size_t stride = (size_t)gridDim.x * blockDim.x;     // multiple of 128
    int c = (int)(tid & 127);
    float s = 0.0f, s2 = 0.0f;
    for (size_t i = tid; i < (size_t)NNODES * HIDDIM; i += stride) {
        float v = acc[i]; s += v; s2 += v * v;
    }
    atomicAdd(&sums[c], s);
    atomicAdd(&sumsq[c], s2);
}

__global__ void bn_finalize_kernel(const float* __restrict__ sums,
                                   const float* __restrict__ sumsq,
                                   const float* __restrict__ bias,
                                   float* __restrict__ mu, float* __restrict__ rstd) {
    int c = threadIdx.x;
    float invN = 1.0f / (float)NNODES;
    float bv = bias[c];
    float mean = sums[c] * invN + bv;                    // mean of (acc+bias)
    float ex2  = (sumsq[c] + 2.0f * bv * sums[c]) * invN + bv * bv;
    float var  = ex2 - mean * mean;                      // biased var (ddof=0)
    mu[c] = mean;
    rstd[c] = rsqrtf(var + 1e-5f);
}

// h = ELU((acc+bias - mu)*rstd*g + b) [+ h residual]
__global__ void __launch_bounds__(256)
bn_elu_kernel(const float* __restrict__ acc, const float* __restrict__ bias,
              const float* __restrict__ mu, const float* __restrict__ rstd,
              const float* __restrict__ gg, const float* __restrict__ bb,
              float* __restrict__ h, int residual) {
    size_t i = blockIdx.x * (size_t)blockDim.x + threadIdx.x;
    if (i >= (size_t)NNODES * HIDDIM) return;
    int c = (int)(i & 127);
    float v = acc[i] + bias[c];
    float y = (v - mu[c]) * rstd[c] * gg[c] + bb[c];
    y = (y > 0.0f) ? y : (__expf(y) - 1.0f);             // ELU(alpha=1)
    h[i] = residual ? (y + h[i]) : y;
}

// ===========================================================================
// Mean pooling per graph (batch is sorted but scatter-add matches reference)
// ===========================================================================
__global__ void pool_count_kernel(const int* __restrict__ batch,
                                  float* __restrict__ counts) {
    int n = blockIdx.x * blockDim.x + threadIdx.x;
    if (n < NNODES) atomicAdd(&counts[batch[n]], 1.0f);
}

__global__ void __launch_bounds__(256)
pool_sum_kernel(const float* __restrict__ h, const int* __restrict__ batch,
                float* __restrict__ emb) {
    size_t i = blockIdx.x * (size_t)blockDim.x + threadIdx.x;
    if (i >= (size_t)NNODES * HIDDIM) return;
    int n = (int)(i >> 7), c = (int)(i & 127);
    atomicAdd(&emb[(size_t)batch[n] * HIDDIM + c], h[i]);
}

__global__ void pool_div_kernel(float* __restrict__ emb,
                                const float* __restrict__ counts) {
    int i = blockIdx.x * blockDim.x + threadIdx.x;
    if (i >= NGRAPH * HIDDIM) return;
    emb[i] /= fmaxf(counts[i >> 7], 1.0f);
}

// ===========================================================================
// Task heads: out[g,t] = relu([emb;fp_sel] @ tw1[t] + tb1[t]) @ tw2[t] + tb2[t]
// One 128-thread block per (g,t); fused vector staged in LDS.
// ===========================================================================
__global__ void __launch_bounds__(128)
heads_kernel(const float* __restrict__ emb, const float* __restrict__ fp,
             const int* __restrict__ fp_idx, const float* __restrict__ tw1,
             const float* __restrict__ tb1, const float* __restrict__ tw2,
             const float* __restrict__ tb2, float* __restrict__ out) {
    __shared__ float fused[HIDDIM + KSEL];
    __shared__ float red[128];
    int g = blockIdx.x / NTASK, t = blockIdx.x % NTASK;
    int c = threadIdx.x;
    fused[c] = emb[(size_t)g * HIDDIM + c];
    if (c < KSEL)
        fused[HIDDIM + c] = fp[(size_t)g * FPBITS + fp_idx[t * KSEL + c]];
    __syncthreads();
    float s = tb1[t * HIDDIM + c];
    #pragma unroll 8
    for (int k = 0; k < HIDDIM + KSEL; ++k)
        s += fused[k] * tw1[((size_t)t * (HIDDIM + KSEL) + k) * HIDDIM + c];
    s = fmaxf(s, 0.0f);
    red[c] = s * tw2[t * HIDDIM + c];
    __syncthreads();
    for (int o = 64; o > 0; o >>= 1) {
        if (c < o) red[c] += red[c + o];
        __syncthreads();
    }
    if (c == 0) out[g * NTASK + t] = red[0] + tb2[t];
}

// ===========================================================================
extern "C" void kernel_launch(void* const* d_in, const int* in_sizes, int n_in,
                              void* d_out, int out_size, void* d_ws, size_t ws_size,
                              hipStream_t stream) {
    // ---- inputs in setup_inputs() order ----
    const float* x       = (const float*)d_in[0];
    const int*   eidx    = (const int*)  d_in[1];
    const float* eattr   = (const float*)d_in[2];
    const int*   batch   = (const int*)  d_in[3];
    const float* morgan  = (const float*)d_in[4];
    const int*   fp_idx  = (const int*)  d_in[5];
    const float* Wl0 = (const float*)d_in[6];   const float* bl0 = (const float*)d_in[7];
    const float* Wr0 = (const float*)d_in[8];   const float* br0 = (const float*)d_in[9];
    const float* We0 = (const float*)d_in[10];  const float* att0= (const float*)d_in[11];
    const float* bias0=(const float*)d_in[12];  const float* g0  = (const float*)d_in[13];
    const float* b0  = (const float*)d_in[14];
    const float* Wl  = (const float*)d_in[15];  const float* bl  = (const float*)d_in[16];
    const float* Wr  = (const float*)d_in[17];  const float* br  = (const float*)d_in[18];
    const float* We  = (const float*)d_in[19];  const float* attL= (const float*)d_in[20];
    const float* biasc=(const float*)d_in[21];  const float* gL  = (const float*)d_in[22];
    const float* bL  = (const float*)d_in[23];
    const float* tw1 = (const float*)d_in[24];  const float* tb1 = (const float*)d_in[25];
    const float* tw2 = (const float*)d_in[26];  const float* tb2 = (const float*)d_in[27];
    float* out = (float*)d_out;

    const int* srcp = eidx;
    const int* dstp = eidx + NEDGES;

    // ---- workspace carve (256B aligned) ----
    char* ws = (char*)d_ws;
    auto carve = [&](size_t bytes) -> void* {
        void* p = (void*)ws;
        ws += (bytes + 255) & ~(size_t)255;
        return p;
    };
    float*    h     = (float*)   carve((size_t)NNODES * HIDDIM * 4);
    float*    xl    = (float*)   carve((size_t)NNODES * HIDDIM * 4);
    float*    xr    = (float*)   carve((size_t)NNODES * HIDDIM * 4);
    float*    acc   = xr;  // xr is dead after edge_score; reuse as accumulator
    float*    score = (float*)   carve((size_t)NEDGES * NHEADS * 4);
    unsigned* m_u   = (unsigned*)carve((size_t)NNODES * NHEADS * 4);
    float*    denom = (float*)   carve((size_t)NNODES * NHEADS * 4);
    float*    stats = (float*)   carve(256 * 4);   // [sums(128) | sumsq(128)]
    float*    mu    = (float*)   carve(128 * 4);
    float*    rstd  = (float*)   carve(128 * 4);
    float*    counts= (float*)   carve(NGRAPH * 4);
    float*    emb   = (float*)   carve((size_t)NGRAPH * HIDDIM * 4);

    const int gemmGrid  = (NNODES + 127) / 128;
    const int edgeGrid  = (NEDGES + 7) / 8;
    const int ehGrid    = (NEDGES * NHEADS + 255) / 256;
    const int nodeGrid  = ((size_t)NNODES * HIDDIM + 255) / 256;

    for (int i = 0; i < NLAYER; ++i) {
        const float* A    = (i == 0) ? x : h;
        const int    K    = (i == 0) ? FNODE : HIDDIM;
        const float* Wl_i = (i == 0) ? Wl0  : Wl   + (size_t)(i-1) * HIDDIM * HIDDIM;
        const float* bl_i = (i == 0) ? bl0  : bl   + (size_t)(i-1) * HIDDIM;
        const float* Wr_i = (i == 0) ? Wr0  : Wr   + (size_t)(i-1) * HIDDIM * HIDDIM;
        const float* br_i = (i == 0) ? br0  : br   + (size_t)(i-1) * HIDDIM;
        const float* We_i = (i == 0) ? We0  : We   + (size_t)(i-1) * FEDGE * HIDDIM;
        const float* at_i = (i == 0) ? att0 : attL + (size_t)(i-1) * HIDDIM;
        const float* bi_i = (i == 0) ? bias0: biasc+ (size_t)(i-1) * HIDDIM;
        const float* g_i  = (i == 0) ? g0   : gL   + (size_t)(i-1) * HIDDIM;
        const float* b_i  = (i == 0) ? b0   : bL   + (size_t)(i-1) * HIDDIM;

        // node projections (WMMA)
        gemm_bf16_wmma_kernel<<<gemmGrid, 256, 0, stream>>>(A, Wl_i, bl_i, xl, NNODES, K);
        gemm_bf16_wmma_kernel<<<gemmGrid, 256, 0, stream>>>(A, Wr_i, br_i, xr, NNODES, K);

        // segment softmax max
        hipMemsetAsync(m_u,   0, (size_t)NNODES * NHEADS * 4, stream);
        hipMemsetAsync(denom, 0, (size_t)NNODES * NHEADS * 4, stream);
        edge_score_kernel<<<edgeGrid, 256, 0, stream>>>(srcp, dstp, eattr, We_i,
                                                        at_i, xl, xr, score, m_u);
        // xr dead now -> zero it for use as the scatter accumulator
        hipMemsetAsync(acc, 0, (size_t)NNODES * HIDDIM * 4, stream);
        edge_exp_kernel<<<ehGrid, 256, 0, stream>>>(dstp, m_u, score, denom);
        edge_scatter_kernel<<<edgeGrid, 256, 0, stream>>>(srcp, dstp, score, denom, xl, acc);

        // BatchNorm(train stats) + ELU (+ residual for layers > 0)
        hipMemsetAsync(stats, 0, 256 * 4, stream);
        bn_stats_kernel<<<512, 256, 0, stream>>>(acc, stats, stats + 128);
        bn_finalize_kernel<<<1, 128, 0, stream>>>(stats, stats + 128, bi_i, mu, rstd);
        bn_elu_kernel<<<nodeGrid, 256, 0, stream>>>(acc, bi_i, mu, rstd, g_i, b_i,
                                                    h, (i > 0) ? 1 : 0);
    }

    // mean pool per graph
    hipMemsetAsync(counts, 0, NGRAPH * 4, stream);
    hipMemsetAsync(emb,    0, (size_t)NGRAPH * HIDDIM * 4, stream);
    pool_count_kernel<<<(NNODES + 255) / 256, 256, 0, stream>>>(batch, counts);
    pool_sum_kernel<<<nodeGrid, 256, 0, stream>>>(h, batch, emb);
    pool_div_kernel<<<(NGRAPH * HIDDIM + 255) / 256, 256, 0, stream>>>(emb, counts);

    // task heads
    heads_kernel<<<NGRAPH * NTASK, 128, 0, stream>>>(emb, morgan, fp_idx,
                                                     tw1, tb1, tw2, tb2, out);
}
